// DIONEMA_18021682774612
// MI455X (gfx1250) — matrix operand order repaired and checked by reading, not compiled, not run on hardware
//
#include <hip/hip_runtime.h>
#include <math.h>

// ---------------------------------------------------------------------------
// MI455X (gfx1250) clustering kernel.
// Memory-bound budget: ~280MB @ 23.3 TB/s ~= 12us. x (134MB) fits in 192MB L2,
// so the scatter pass re-reads x from L2. Matmul (34.4 GFLOP) uses
// v_wmma_f32_16x16x32_f16 (inputs are L2-normalized, |v|<=1 -> f16-safe).
// Streaming passes use b128-per-lane VMEM to stay under the vmem issue-rate
// ceiling; centroid tiles are staged global->LDS with the CDNA5 async path
// (global_load_async_to_lds_b128 + s_wait_asynccnt), double-buffered so the
// copy overlaps the WMMA work with one barrier per tile.
// ---------------------------------------------------------------------------

typedef _Float16 f16;
typedef __attribute__((ext_vector_type(8)))  _Float16 v8h;
typedef __attribute__((ext_vector_type(16))) _Float16 v16h;
typedef __attribute__((ext_vector_type(8)))  float    v8f;

#define Bb     64
#define Dd     128
#define HWp    4096                 // H*W
#define Nn     (Bb*HWp)             // 262144 pixels
#define Kk     512
#define TILE   128                  // pixels per block
#define NT16   (Kk/16)              // 32 centroid tiles
#define LXH    136                  // padded half-stride (272B, 16B aligned)
#define MARGIN 0.005f
#define EPSN   1e-12f

// ---------------------------------------------------------------------------
// Stage 0: normalize centroids -> f16 [K][D] + |c_norm|^2
// ---------------------------------------------------------------------------
__global__ __launch_bounds__(128)
void DIONEMA_centroid_prep(const float* __restrict__ cent,
                           f16* __restrict__ cent_h,
                           float* __restrict__ cn2) {
  __shared__ float red[128];
  __shared__ float s_inv;
  const int k = blockIdx.x;
  const int t = threadIdx.x;
  const float c = cent[k * Dd + t];
  red[t] = c * c;
  __syncthreads();
  for (int s = 64; s > 0; s >>= 1) {
    if (t < s) red[t] += red[t + s];
    __syncthreads();
  }
  if (t == 0) {
    const float ss  = red[0];
    const float inv = 1.0f / fmaxf(sqrtf(ss), EPSN);
    s_inv = inv;
    cn2[k] = ss * inv * inv;      // == |c_norm|^2 (faithful to reference)
  }
  __syncthreads();
  cent_h[k * Dd + t] = (f16)(c * s_inv);
}

// ---------------------------------------------------------------------------
// Main fused kernel: normalize + WMMA distances + top-2/argmin + scatter.
// Block: 256 threads (8 waves). Each wave owns a 16-pixel M-tile (A-fragments
// resident in VGPRs); B tiles double-buffered global->LDS via async copy.
// Grid: N/TILE = 2048 blocks.
// ---------------------------------------------------------------------------
__global__ __launch_bounds__(256)
void DIONEMA_cluster_main(const float* __restrict__ x,
                          const f16* __restrict__ cent_h,
                          const float* __restrict__ cn2g,
                          float* __restrict__ outf,
                          float* __restrict__ csums,
                          float* __restrict__ counts,
                          int* __restrict__ idx_out) {
  __shared__ f16   lds_xh[TILE * LXH];      // normalized pixels, f16, padded
  __shared__ f16   lds_B[2][16 * LXH];      // double-buffered centroid tile
  __shared__ float lds_ss[TILE * 9];        // pad-9: conflict-free reduce
  __shared__ float lds_inv[TILE];
  __shared__ float lds_xn2[TILE];
  __shared__ int   res_idx[TILE];
  __shared__ int   res_above[TILE];

  const int t   = threadIdx.x;
  const int n0  = blockIdx.x * TILE;        // first pixel of tile (one b)
  const int b   = n0 / HWp;
  const int hw0 = n0 % HWp;
  const size_t xbase = (size_t)b * Dd * HWp + hw0;

  // float4 streaming mapping: q = pixel quad (pixels q*4..q*4+3, contiguous),
  // wrow = feature phase; lane-level b128, fully coalesced (512B/instr/wave).
  const int q    = t & 31;
  const int wrow = t >> 5;

  // ---- Issue async copy of centroid tile 0 (overlaps the whole pass 1) ----
  const int cn = t >> 4;                    // centroid row within tile
  const int dh = (t & 15) * 8;              // half offset within row
  const unsigned ldsa[2] = {
    (unsigned)(size_t)&lds_B[0][cn * LXH + dh],
    (unsigned)(size_t)&lds_B[1][cn * LXH + dh]
  };
  {
    const f16* g = cent_h + t * 8;
    asm volatile("global_load_async_to_lds_b128 %0, %1, off"
                 :: "v"(ldsa[0]), "v"(g) : "memory");
  }

  // ---- Pass 1a: per-pixel sum of squares (b128 streaming read) ----
  {
    float4 a4 = make_float4(0.f, 0.f, 0.f, 0.f);
#pragma unroll 4
    for (int i = 0; i < 16; ++i) {
      const int d = wrow + 8 * i;
      const float4 v = *(const float4*)(x + xbase + (size_t)d * HWp + q * 4);
      a4.x += v.x * v.x; a4.y += v.y * v.y;
      a4.z += v.z * v.z; a4.w += v.w * v.w;
    }
    lds_ss[(q * 4 + 0) * 9 + wrow] = a4.x;
    lds_ss[(q * 4 + 1) * 9 + wrow] = a4.y;
    lds_ss[(q * 4 + 2) * 9 + wrow] = a4.z;
    lds_ss[(q * 4 + 3) * 9 + wrow] = a4.w;
  }
  __syncthreads();
  if (t < TILE) {
    float ss = 0.f;
#pragma unroll
    for (int w = 0; w < 8; ++w) ss += lds_ss[t * 9 + w];
    const float inv = 1.0f / fmaxf(sqrtf(ss), EPSN);
    lds_inv[t] = inv;
    lds_xn2[t] = ss * inv * inv;            // == |norm_x|^2 (faithful)
  }
  __syncthreads();

  // ---- Pass 1b: write normalized output (b128) + build f16 A tile ----
  {
    const float4 inv4 = make_float4(lds_inv[q * 4 + 0], lds_inv[q * 4 + 1],
                                    lds_inv[q * 4 + 2], lds_inv[q * 4 + 3]);
#pragma unroll 4
    for (int i = 0; i < 16; ++i) {
      const int d = wrow + 8 * i;
      const size_t off = xbase + (size_t)d * HWp + q * 4;
      const float4 v = *(const float4*)(x + off);   // L2 hit (just streamed)
      float4 nx;
      nx.x = v.x * inv4.x; nx.y = v.y * inv4.y;
      nx.z = v.z * inv4.z; nx.w = v.w * inv4.w;
      *(float4*)(outf + off) = nx;                  // b128 store
      lds_xh[(q * 4 + 0) * LXH + d] = (f16)nx.x;
      lds_xh[(q * 4 + 1) * LXH + d] = (f16)nx.y;
      lds_xh[(q * 4 + 2) * LXH + d] = (f16)nx.z;
      lds_xh[(q * 4 + 3) * LXH + d] = (f16)nx.w;
    }
  }
  asm volatile("s_wait_asynccnt 0x0" ::: "memory");  // tile 0 landed
  __syncthreads();

  // ---- Pass 2: WMMA distances, per-lane running top-2 ----
  const int lane    = t & 31;
  const int wave    = t >> 5;               // 8 waves == 8 M-tiles
  const int mt      = wave;
  const int halfsel = lane >> 4;            // 0: lanes 0-15, 1: lanes 16-31
  const int col     = lane & 15;

  // Resident A fragments: lane holds row (lane&15); halves per 16-bit A
  // layout: lanes<16 K{0..7,16..23}, lanes>=16 K{8..15,24..31} per K-chunk.
  v16h afr[4];
#pragma unroll
  for (int kc = 0; kc < 4; ++kc) {
    const f16* ap = &lds_xh[(mt * 16 + col) * LXH + kc * 32 + halfsel * 8];
    const v8h lo = *(const v8h*)ap;           // ds_load_b128
    const v8h hi = *(const v8h*)(ap + 16);    // ds_load_b128
    afr[kc] = __builtin_shufflevector(lo, hi, 0,1,2,3,4,5,6,7,8,9,10,11,12,13,14,15);
  }

  float xn2r[8];
#pragma unroll
  for (int r = 0; r < 8; ++r)
    xn2r[r] = lds_xn2[mt * 16 + halfsel * 8 + r];   // LDS broadcast

  float d1[8], d2[8];
  int   i1[8];
#pragma unroll
  for (int r = 0; r < 8; ++r) { d1[r] = 3.4e38f; d2[r] = 3.4e38f; i1[r] = 0; }

  for (int nt = 0; nt < NT16; ++nt) {
    const int buf = nt & 1;
    // Kick async copy of tile nt+1 into the other buffer. Safe: its previous
    // readers (iteration nt-1) are behind the barrier that ended iteration
    // nt-1. Copy overlaps the fragment loads + WMMAs below.
    if (nt + 1 < NT16) {
      const f16* g = cent_h + (size_t)(nt + 1) * 16 * Dd + t * 8;
      asm volatile("global_load_async_to_lds_b128 %0, %1, off"
                   :: "v"(ldsa[buf ^ 1]), "v"(g) : "memory");
    }

    // B fragment: lane holds column (lane&15); lanes<16 K{0..7,16..23}, >=16 K{8..15,24..31}.
    v8f accv = {0.f, 0.f, 0.f, 0.f, 0.f, 0.f, 0.f, 0.f};
#pragma unroll
    for (int kc = 0; kc < 4; ++kc) {
      const f16* bp = &lds_B[buf][col * LXH + kc * 32 + halfsel * 8];
      const v8h lo = *(const v8h*)bp;
      const v8h hi = *(const v8h*)(bp + 16);
      const v16h bfr = __builtin_shufflevector(lo, hi, 0,1,2,3,4,5,6,7,8,9,10,11,12,13,14,15);
      accv = __builtin_amdgcn_wmma_f32_16x16x32_f16(
          false, afr[kc], false, bfr, (short)0, accv, false, false);
    }

    // dist = |x|^2 + |c|^2 - 2 x.c  (exactly the reference expression)
    const float cn2v = cn2g[nt * 16 + col];
    const int   cid  = nt * 16 + col;
#pragma unroll
    for (int r = 0; r < 8; ++r) {
      const float dist   = xn2r[r] + cn2v - 2.0f * accv[r];
      const bool  better = dist < d1[r];
      d2[r] = better ? d1[r] : fminf(d2[r], dist);
      i1[r] = better ? cid   : i1[r];
      d1[r] = better ? dist  : d1[r];
    }

    asm volatile("s_wait_asynccnt 0x0" ::: "memory");  // next tile landed
    __syncthreads();                                   // one barrier per tile
  }

  // ---- Cross-lane top-2 merge within each 16-lane group (XOR tree) ----
#pragma unroll
  for (int r = 0; r < 8; ++r) {
    float a1 = d1[r], a2 = d2[r];
    int   ai = i1[r];
    for (int m = 1; m < 16; m <<= 1) {
      const float o1 = __shfl_xor(a1, m, 32);
      const float o2 = __shfl_xor(a2, m, 32);
      const int   oi = __shfl_xor(ai, m, 32);
      const float lo = fminf(a1, o1);
      const float hi = fmaxf(a1, o1);
      ai = (o1 < a1) ? oi : ai;
      a2 = fminf(hi, fminf(a2, o2));   // 2nd smallest of union
      a1 = lo;
    }
    d1[r] = a1; d2[r] = a2; i1[r] = ai;
  }
  if (col == 0) {   // lanes 0 and 16 hold reduced rows {0..7} / {8..15}
#pragma unroll
    for (int r = 0; r < 8; ++r) {
      const int row = mt * 16 + halfsel * 8 + r;
      res_idx[row]   = i1[r];
      res_above[row] = ((d2[r] - d1[r]) > MARGIN) ? 1 : 0;
    }
  }
  __syncthreads();

  // ---- Pass 3: idx, counts, masked per-cluster scatter of RAW features ----
  if (t < TILE) {
    const int cid = res_idx[t];
    idx_out[n0 + t] = cid;
    if (res_above[t]) atomicAdd(&counts[cid], 1.0f);
  }
  {
    const int p  = t & 127;
    const int d0 = t >> 7;
    if (res_above[p]) {
      const int cid = res_idx[p];
#pragma unroll 4
      for (int i = 0; i < 64; ++i) {
        const int d = d0 + 2 * i;
        const float v = x[xbase + (size_t)d * HWp + p];   // L2-resident
        atomicAdd(&csums[cid * Dd + d], v);
      }
    }
  }
}

// ---------------------------------------------------------------------------
extern "C" void kernel_launch(void* const* d_in, const int* in_sizes, int n_in,
                              void* d_out, int out_size, void* d_ws, size_t ws_size,
                              hipStream_t stream) {
  (void)in_sizes; (void)n_in; (void)out_size; (void)ws_size;
  const float* x    = (const float*)d_in[0];   // (64,128,64,64) f32
  const float* cent = (const float*)d_in[1];   // (512,128) f32

  float* outf    = (float*)d_out;                       // 33554432 f32
  float* csums   = outf + (size_t)Bb * Dd * HWp;        // +65536 f32
  float* counts  = csums + (size_t)Kk * Dd;             // +512 f32
  int*   idx_out = (int*)(counts + Kk);                 // +262144 i32

  f16*   cent_h = (f16*)d_ws;                                    // 128 KB
  float* cn2    = (float*)((char*)d_ws + (size_t)Kk * Dd * sizeof(f16));

  // zero the accumulated outputs (poisoned by harness)
  hipMemsetAsync(csums, 0, (size_t)(Kk * Dd + Kk) * sizeof(float), stream);

  DIONEMA_centroid_prep<<<Kk, 128, 0, stream>>>(cent, cent_h, cn2);
  DIONEMA_cluster_main<<<Nn / TILE, 256, 0, stream>>>(
      x, cent_h, cn2, outf, csums, counts, idx_out);
}
